// Attention_2972117369285
// MI455X (gfx1250) — matrix operand order repaired and checked by reading, not compiled, not run on hardware
//
#include <hip/hip_runtime.h>

// Causal MHA forward for B=2, T=2048, D=1024, H=16, DH=64 on gfx1250.
// All matmuls via v_wmma_f32_16x16x32_bf16 (bf16 operands, fp32 accum).
// K/V tiles and out-proj A tiles staged with global_load_async_to_lds_b128.

static constexpr int kB  = 2;
static constexpr int kT  = 2048;
static constexpr int kD  = 1024;
static constexpr int kH  = 16;
static constexpr int kDH = 64;

typedef __attribute__((ext_vector_type(16))) __bf16 v16bf;
typedef __attribute__((ext_vector_type(4)))  __bf16 v4bf;
typedef __attribute__((ext_vector_type(8)))  float  v8f;

#define WMMA_BF16(a, b, c) \
  __builtin_amdgcn_wmma_f32_16x16x32_bf16(false, (a), false, (b), (short)0, (c), false, false)

// ---- gfx1250 async copy: global -> LDS, 16B per lane (ASYNCcnt) ----------
// GVS mode: mem_addr = SGPR base + per-lane 32-bit byte offset.
// LDS destination address = low 32 bits of the generic shared pointer
// (flat LDS aperture: ADDR[31:0] is the LDS offset).
__device__ __forceinline__ void async_copy_b128(unsigned lds_addr,
                                                const void* sbase,
                                                unsigned voff) {
  asm volatile("global_load_async_to_lds_b128 %0, %1, %2"
               :: "v"(lds_addr), "v"(voff), "s"(sbase)
               : "memory");
}
__device__ __forceinline__ void wait_async0() {
  asm volatile("s_wait_asynccnt 0x0" ::: "memory");
}
__device__ __forceinline__ unsigned lds_addr_of(const void* p) {
  return (unsigned)(unsigned long long)p;
}

// ---- Fragment loaders (contiguous k = 16*half + s order, wave32) -------------
// A 16x32 bf16: lane holds row m=lane&15; K pairs:
//   lanes 0-15 : V0-7 -> K 0..15
//   lanes 16-31: V0-7 -> K 16..31
__device__ __forceinline__ v16bf frag_a(const __bf16* p, int ldr, int lane) {
  const int m = lane & 15, half = lane >> 4;
  v16bf a;
#pragma unroll
  for (int r = 0; r < 8; ++r) {
    const int k = (half << 4) + 2 * r;
    a[2 * r]     = p[m * ldr + k];
    a[2 * r + 1] = p[m * ldr + k + 1];
  }
  return a;
}

// B 32x16 bf16 from row-major [k][n] storage: lane holds col n=lane&15;
// lanes 0-15 K=0..15 across V0-7 (pairs), lanes 16-31 K=16..31.
__device__ __forceinline__ v16bf frag_b(const __bf16* p, int ldr, int lane) {
  const int n = lane & 15, half = lane >> 4;
  v16bf b;
#pragma unroll
  for (int r = 0; r < 8; ++r) {
    const int k = (half << 4) + 2 * r;
    b[2 * r]     = p[k * ldr + n];
    b[2 * r + 1] = p[(k + 1) * ldr + n];
  }
  return b;
}

// Same B fragment but from transposed storage [n][k] (K pairs contiguous).
__device__ __forceinline__ v16bf frag_bt(const __bf16* p, int ldr, int lane) {
  const int n = lane & 15, half = lane >> 4;
  v16bf b;
#pragma unroll
  for (int r = 0; r < 8; ++r) {
    const int k = (half << 4) + 2 * r;
    b[2 * r]     = p[n * ldr + k];
    b[2 * r + 1] = p[n * ldr + k + 1];
  }
  return b;
}

// ---- Kernel 1: fused QKV projection -------------------------------------
// C[4096,1024] = x @ W + b ; blockIdx.z selects {q,k,v}. Output bf16,
// head-split [B,H,T,DH]; q gets the DH^-0.5 scale folded in. BK = 64.
__global__ __launch_bounds__(256) void qkv_proj_kernel(
    const float* __restrict__ x,
    const float* __restrict__ Wq, const float* __restrict__ bq,
    const float* __restrict__ Wk, const float* __restrict__ bk,
    const float* __restrict__ Wv, const float* __restrict__ bv,
    __bf16* __restrict__ qws, __bf16* __restrict__ kws, __bf16* __restrict__ vws) {
  const int which = blockIdx.z;
  const float* W  = (which == 0) ? Wq : (which == 1) ? Wk : Wv;
  const float* bs = (which == 0) ? bq : (which == 1) ? bk : bv;
  __bf16* dst     = (which == 0) ? qws : (which == 1) ? kws : vws;
  const float scale = (which == 0) ? 0.125f : 1.0f;  // DH^-0.5 = 1/8

  const int m0 = blockIdx.x * 128;
  const int n0 = blockIdx.y * 64;
  const int tid = threadIdx.x, lane = tid & 31, w = tid >> 5;
  const int mw = (w >> 1) * 32, nw = (w & 1) * 32;

  __shared__ __bf16 ldsA[128 * 64];   // x tile,  row-major [m][k]
  __shared__ __bf16 ldsBT[64 * 64];   // W tile, transposed [n][k]

  v8f acc[2][2] = {};
  for (int k0 = 0; k0 < kD; k0 += 64) {
    __syncthreads();
    // x tile: float4 loads, packed bf16x4 stores (2048 float4 / 256 thr).
#pragma unroll
    for (int i = 0; i < 8; ++i) {
      const int u = tid + i * 256;
      const int r = u >> 4, c4 = (u & 15) << 2;
      const float4 f = *(const float4*)&x[(m0 + r) * kD + k0 + c4];
      v4bf pk;
      pk[0] = (__bf16)f.x; pk[1] = (__bf16)f.y;
      pk[2] = (__bf16)f.z; pk[3] = (__bf16)f.w;
      *(v4bf*)&ldsA[r * 64 + c4] = pk;
    }
    // W tile: convert + transpose (scalar; off the WMMA critical path).
#pragma unroll
    for (int i = 0; i < 16; ++i) {
      const int e = tid + i * 256;
      const int kk = e >> 6, n = e & 63;
      ldsBT[n * 64 + kk] = (__bf16)W[(k0 + kk) * kD + n0 + n];
    }
    if (k0 + 64 < kD) {  // hint next stage into cache (global_prefetch_b8)
      __builtin_prefetch(&x[(m0 + (tid >> 1)) * kD + k0 + 64 + ((tid & 1) << 5)], 0, 0);
      __builtin_prefetch(&W[(k0 + 64 + (tid >> 2)) * kD + n0 + ((tid & 3) << 4)], 0, 0);
    }
    __syncthreads();
#pragma unroll
    for (int kc = 0; kc < 64; kc += 32) {
      const v16bf a0 = frag_a(&ldsA[(mw + 0) * 64 + kc], 64, lane);
      const v16bf a1 = frag_a(&ldsA[(mw + 16) * 64 + kc], 64, lane);
      const v16bf b0 = frag_bt(&ldsBT[(nw + 0) * 64 + kc], 64, lane);
      const v16bf b1 = frag_bt(&ldsBT[(nw + 16) * 64 + kc], 64, lane);
      acc[0][0] = WMMA_BF16(a0, b0, acc[0][0]);
      acc[0][1] = WMMA_BF16(a0, b1, acc[0][1]);
      acc[1][0] = WMMA_BF16(a1, b0, acc[1][0]);
      acc[1][1] = WMMA_BF16(a1, b1, acc[1][1]);
    }
  }
  const int half = lane >> 4, nn = lane & 15;
#pragma unroll
  for (int i = 0; i < 2; ++i)
#pragma unroll
    for (int j = 0; j < 2; ++j)
#pragma unroll
      for (int r = 0; r < 8; ++r) {
        const int gm = m0 + mw + i * 16 + half * 8 + r;
        const int gn = n0 + nw + j * 16 + nn;
        const float val = (acc[i][j][r] + bs[gn]) * scale;
        const int h = gn >> 6, d = gn & 63;       // head split
        const int b = gm >> 11, t = gm & 2047;    // b*2048 + t
        dst[(((b * kH + h) * kT) + t) * kDH + d] = (__bf16)val;
      }
}

// ---- Kernel 2: flash attention ------------------------------------------
// Grid: (T/128, B*H). 256 threads = 8 waves; wave w owns q rows
// qBase + 16w .. +15. K/V tiles (32 keys) double-buffered via async-to-LDS.
__global__ __launch_bounds__(256) void attn_kernel(
    const __bf16* __restrict__ qws, const __bf16* __restrict__ kws,
    const __bf16* __restrict__ vws, __bf16* __restrict__ attn) {
  const int bh = blockIdx.y;
  const int qBase = blockIdx.x * 128;
  const int tid = threadIdx.x, lane = tid & 31, w = tid >> 5;
  const int half = lane >> 4, nn = lane & 15;
  const int base = bh * (kT * kDH);
  const __bf16* qg = qws + base;
  const __bf16* kg = kws + base;
  const __bf16* vg = vws + base;

  __shared__ __align__(16) __bf16 ldsK[2][32 * 64];  // [buf][key][d]
  __shared__ __align__(16) __bf16 ldsV[2][32 * 64];
  __shared__ __bf16 ldsP[8][16 * 32];                // per-wave P patch

  const unsigned ldsKbase = lds_addr_of(&ldsK[0][0]);
  const unsigned ldsVbase = lds_addr_of(&ldsV[0][0]);
  const unsigned voff = (unsigned)(tid << 4);        // 16B per thread

  const int qRow0 = qBase + w * 16;
  // Wave-private Q fragments: 16 rows x 64 dims = two 16x32 A-frags.
  const v16bf aq0 = frag_a(qg + qRow0 * kDH, kDH, lane);
  const v16bf aq1 = frag_a(qg + qRow0 * kDH + 32, kDH, lane);

  v8f o[4] = {};
  float mrow[8], lrow[8];
#pragma unroll
  for (int r = 0; r < 8; ++r) { mrow[r] = -3.0e38f; lrow[r] = 0.0f; }

  const int kEnd = qBase + 128;          // causal upper bound, uniform per WG
  // Prologue: async-stage tile 0 into buffer 0.
  async_copy_b128(ldsKbase + voff, kg, voff);
  async_copy_b128(ldsVbase + voff, vg, voff);

  for (int kt = 0; kt < kEnd; kt += 32) {
    const int buf = (kt >> 5) & 1;
    wait_async0();                       // own async writes landed
    __syncthreads();                     // publish tile to all waves
    if (kt + 32 < kEnd) {                // overlap next tile with compute
      const unsigned boff = (unsigned)((buf ^ 1) << 12);
      async_copy_b128(ldsKbase + boff + voff, kg + (kt + 32) * kDH, voff);
      async_copy_b128(ldsVbase + boff + voff, vg + (kt + 32) * kDH, voff);
    }
    const __bf16* Kt = &ldsK[buf][0];
    const __bf16* Vt = &ldsV[buf][0];

    // S = q (16x64) @ K^T (64x32): two 16x16 n-tiles, K accumulated 32+32.
    v8f s[2];
#pragma unroll
    for (int j = 0; j < 2; ++j) {
      v8f c = {};
      const v16bf bk0 = frag_bt(&Kt[(j * 16) * 64], 64, lane);
      const v16bf bk1 = frag_bt(&Kt[(j * 16) * 64 + 32], 64, lane);
      c = WMMA_BF16(aq0, bk0, c);
      c = WMMA_BF16(aq1, bk1, c);
      s[j] = c;
    }

    // Causal mask + online softmax (row = half*8 + r, 16 lanes per row).
    float p0[8], p1[8];
#pragma unroll
    for (int r = 0; r < 8; ++r) {
      const int qrow = qRow0 + half * 8 + r;
      float s0 = (kt + nn <= qrow) ? s[0][r] : -3.0e38f;
      float s1 = (kt + 16 + nn <= qrow) ? s[1][r] : -3.0e38f;
      float rm = fmaxf(s0, s1);
#pragma unroll
      for (int m = 1; m < 16; m <<= 1) rm = fmaxf(rm, __shfl_xor(rm, m, 32));
      const float newm = fmaxf(mrow[r], rm);
      const float alpha = __expf(mrow[r] - newm);
      p0[r] = __expf(s0 - newm);
      p1[r] = __expf(s1 - newm);
      float rs = p0[r] + p1[r];
#pragma unroll
      for (int m = 1; m < 16; m <<= 1) rs += __shfl_xor(rs, m, 32);
      lrow[r] = lrow[r] * alpha + rs;
      mrow[r] = newm;
#pragma unroll
      for (int d = 0; d < 4; ++d) o[d][r] *= alpha;
    }

    // Re-layout P (C-frag -> A-frag) through this wave's LDS patch.
#pragma unroll
    for (int r = 0; r < 8; ++r) {
      const int m = half * 8 + r;
      ldsP[w][m * 32 + nn]      = (__bf16)p0[r];
      ldsP[w][m * 32 + 16 + nn] = (__bf16)p1[r];
    }
    const v16bf aP = frag_a(&ldsP[w][0], 32, lane);
#pragma unroll
    for (int d = 0; d < 4; ++d) {
      const v16bf bv = frag_b(&Vt[d * 16], 64, lane);  // V[k][d*16+n]
      o[d] = WMMA_BF16(aP, bv, o[d]);
    }
  }

  // Normalize and store bf16 into attn workspace [B,T,D] (heads re-merged).
  const int b = bh >> 4, h = bh & 15;
#pragma unroll
  for (int d = 0; d < 4; ++d)
#pragma unroll
    for (int r = 0; r < 8; ++r) {
      const int qrow = qRow0 + half * 8 + r;
      const int col = h * kDH + d * 16 + nn;
      attn[(b * kT + qrow) * kD + col] = (__bf16)(o[d][r] / lrow[r]);
    }
}

// ---- Kernel 3: output projection (bf16 A via async-to-LDS, fp32 out) ----
__global__ __launch_bounds__(256) void out_proj_kernel(
    const __bf16* __restrict__ a, const float* __restrict__ Wp,
    const float* __restrict__ bp, float* __restrict__ out) {
  const int m0 = blockIdx.x * 128;
  const int n0 = blockIdx.y * 64;
  const int tid = threadIdx.x, lane = tid & 31, w = tid >> 5;
  const int mw = (w >> 1) * 32, nw = (w & 1) * 32;

  __shared__ __align__(16) __bf16 ldsA[128 * 64];  // [m][k]
  __shared__ __bf16 ldsBT[64 * 64];                // [n][k]
  const unsigned ldsAbase = lds_addr_of(&ldsA[0]);

  v8f acc[2][2] = {};
  for (int k0 = 0; k0 < kD; k0 += 64) {
    __syncthreads();
    // A tile: 128x64 bf16 = 1024 x 16B, async copied (4 per thread).
    {
      const __bf16* gbase = a + m0 * kD + k0;      // uniform SGPR base
#pragma unroll
      for (int i = 0; i < 4; ++i) {
        const int u = tid + i * 256;               // 16B chunk index
        const int r = u >> 3, part = u & 7;
        const unsigned goff = (unsigned)(r * kD * 2 + part * 16);
        async_copy_b128(ldsAbase + (unsigned)(u << 4), gbase, goff);
      }
    }
    // W tile: convert + transpose.
#pragma unroll
    for (int i = 0; i < 16; ++i) {
      const int e = tid + i * 256;
      const int kk = e >> 6, n = e & 63;
      ldsBT[n * 64 + kk] = (__bf16)Wp[(k0 + kk) * kD + n0 + n];
    }
    if (k0 + 64 < kD)
      __builtin_prefetch(&Wp[(k0 + 64 + (tid >> 2)) * kD + n0 + ((tid & 3) << 4)], 0, 0);
    wait_async0();
    __syncthreads();
#pragma unroll
    for (int kc = 0; kc < 64; kc += 32) {
      const v16bf a0 = frag_a(&ldsA[(mw + 0) * 64 + kc], 64, lane);
      const v16bf a1 = frag_a(&ldsA[(mw + 16) * 64 + kc], 64, lane);
      const v16bf b0 = frag_bt(&ldsBT[(nw + 0) * 64 + kc], 64, lane);
      const v16bf b1 = frag_bt(&ldsBT[(nw + 16) * 64 + kc], 64, lane);
      acc[0][0] = WMMA_BF16(a0, b0, acc[0][0]);
      acc[0][1] = WMMA_BF16(a0, b1, acc[0][1]);
      acc[1][0] = WMMA_BF16(a1, b0, acc[1][0]);
      acc[1][1] = WMMA_BF16(a1, b1, acc[1][1]);
    }
  }
  const int half = lane >> 4, nn = lane & 15;
#pragma unroll
  for (int i = 0; i < 2; ++i)
#pragma unroll
    for (int j = 0; j < 2; ++j)
#pragma unroll
      for (int r = 0; r < 8; ++r) {
        const int gm = m0 + mw + i * 16 + half * 8 + r;
        const int gn = n0 + nw + j * 16 + nn;
        out[gm * kD + gn] = acc[i][j][r] + bp[gn];
      }
}

// ---- Host launcher -------------------------------------------------------
extern "C" void kernel_launch(void* const* d_in, const int* in_sizes, int n_in,
                              void* d_out, int out_size, void* d_ws, size_t ws_size,
                              hipStream_t stream) {
  (void)in_sizes; (void)n_in; (void)out_size; (void)ws_size;
  const float* x  = (const float*)d_in[0];
  const float* Wq = (const float*)d_in[1];
  const float* bq = (const float*)d_in[2];
  const float* Wk = (const float*)d_in[3];
  const float* bk = (const float*)d_in[4];
  const float* Wv = (const float*)d_in[5];
  const float* bv = (const float*)d_in[6];
  const float* Wp = (const float*)d_in[7];
  const float* bp = (const float*)d_in[8];
  float* out = (float*)d_out;

  // Workspace: q,k,v head-split [B,H,T,DH] bf16 + attn [B,T,D] bf16 = 32 MB.
  const size_t HEADS = (size_t)kB * kH * kT * kDH;
  __bf16* qws   = (__bf16*)d_ws;
  __bf16* kws   = qws + HEADS;
  __bf16* vws   = kws + HEADS;
  __bf16* attnw = vws + HEADS;

  qkv_proj_kernel<<<dim3(32, 16, 3), 256, 0, stream>>>(x, Wq, bq, Wk, bk, Wv, bv,
                                                       qws, kws, vws);
  attn_kernel<<<dim3(16, 32), 256, 0, stream>>>(qws, kws, vws, attnw);
  out_proj_kernel<<<dim3(32, 16), 256, 0, stream>>>(attnw, Wp, bp, out);
}
